// SparseCodingLayer_85134841741545
// MI455X (gfx1250) — matrix operand order repaired and checked by reading, not compiled, not run on hardware
//
#include <hip/hip_runtime.h>
#include <hip/hip_bf16.h>
#include <math.h>
#include <stdint.h>

// ---------------------------------------------------------------------------
// Types for CDNA5 WMMA (wave32):  D(16x16 f32) = A(16x32 bf16) x B(32x16 bf16) + C
// ---------------------------------------------------------------------------
typedef __attribute__((ext_vector_type(16))) __bf16 v16bf;
typedef __attribute__((ext_vector_type(8)))  __bf16 v8bf;
typedef __attribute__((ext_vector_type(8)))  float  v8f;
typedef __attribute__((ext_vector_type(4)))  int    v4i;

// Problem constants (from reference):
//   X: (8*256=2048, 1200) fp32, D: (1200, 2400) fp32
#define M_ROWS   2048
#define DS       1200          // dim_signal
#define DD       2400          // dim_dict
#define DS_PAD   1216          // 1200 padded to multiple of 32
#define DD_PAD   2432          // 2400 padded to multiple of 64
#define ITER_FISTA    16
#define ITER_EIG      20
#define COUPLE        0.1f

#define LDSSTR   40            // LDS row stride in halves (32 + 8 pad -> bank-conflict free)

// CDNA5 async global->LDS copy path (ASYNCcnt), if the toolchain exposes it.
#if defined(__AMDGCN__) && \
    __has_builtin(__builtin_amdgcn_global_load_async_to_lds_b128) && \
    __has_builtin(__builtin_amdgcn_s_wait_asynccnt)
#define USE_ASYNC_LDS 1
#define AS1 __attribute__((address_space(1)))
#define AS3 __attribute__((address_space(3)))
// Param types per clang: (v4i AS1* gsrc, v4i AS3* ldst, imm offset, imm cpol).
// Integer casts sidestep C-style addrspacecast restrictions; LDS offset is the
// low 32 bits of the flat address (ISA aperture rule), AS3 pointers are 32-bit.
__device__ __forceinline__ void async_cp16(void* lds, const void* gmem) {
  __builtin_amdgcn_global_load_async_to_lds_b128(
      (AS1 v4i*)(uintptr_t)gmem,
      (AS3 v4i*)(uint32_t)(uintptr_t)lds,
      0, 0);
}
#else
#define USE_ASYNC_LDS 0
#endif

// ---------------------------------------------------------------------------
// Tiled NT GEMM with WMMA:  C[m,n] = sum_k A[m,k] * Bt[n,k]   (bf16 in, f32 acc)
// Block: 256 threads = 8 waves, tile 128(M) x 64(N), BK = 32, double-buffered LDS.
// MODE 1: residual   -> OutBf[m,n] = acc - X[m,n]           (bf16, ld DS_PAD)
// MODE 2: grad+FISTA -> fused soft-threshold / momentum update of (z, y)
// MODE 3: final      -> OutF[m,n] = acc                      (f32, ld DS)
// ---------------------------------------------------------------------------
template<int MODE>
__global__ __launch_bounds__(256)
void gemm_nt_wmma(const __bf16* __restrict__ A,  int lda,
                  const __bf16* __restrict__ Bt, int ldb,
                  int N, int Ksteps,
                  const float*  __restrict__ Xres,   // MODE 1
                  __bf16*       __restrict__ OutBf,  // MODE 1
                  float*        __restrict__ OutF,   // MODE 3
                  const float*  __restrict__ Lp,     // MODE 2
                  float* __restrict__ Zf,  __bf16* __restrict__ Zbf,
                  float* __restrict__ Yf,  __bf16* __restrict__ Ybf,
                  float mu)
{
  __shared__ __bf16 sA[2][128 * LDSSTR];
  __shared__ __bf16 sB[2][ 64 * LDSSTR];

  const int tid    = threadIdx.x;
  const int lane   = tid & 31;
  const int wid    = tid >> 5;
  const int wm     = wid & 3;       // wave M quadrant (32 rows each)
  const int wn     = wid >> 2;      // wave N half     (32 cols each)
  const int blockM = blockIdx.y * 128;
  const int blockN = blockIdx.x * 64;
  const int hsel   = lane >> 4;     // 0 | 1 (lane half)
  const int l15    = lane & 15;

  // staging coordinates: A tile = 512 x (8-half chunks), B tile = 256 chunks
  const int rA0 = tid >> 2,         cA0 = (tid & 3) * 8;            // chunk tid
  const int rA1 = (tid + 256) >> 2, cA1 = ((tid + 256) & 3) * 8;    // chunk tid+256
  const int rB  = tid >> 2,         cB  = (tid & 3) * 8;

  v8f acc[2][2] = {};

#if USE_ASYNC_LDS
  auto issueTile = [&](int ks, int buf) {
    const int k0 = ks * 32;
    async_cp16(&sA[buf][rA0 * LDSSTR + cA0],
               A + (size_t)(blockM + rA0) * lda + (k0 + cA0));
    async_cp16(&sA[buf][rA1 * LDSSTR + cA1],
               A + (size_t)(blockM + rA1) * lda + (k0 + cA1));
    async_cp16(&sB[buf][rB * LDSSTR + cB],
               Bt + (size_t)(blockN + rB) * ldb + (k0 + cB));
  };
  issueTile(0, 0);
#else
  v8bf tA0, tA1, tB;
  auto loadTile = [&](int ks) {
    const int k0 = ks * 32;
    tA0 = *(const v8bf*)(A  + (size_t)(blockM + rA0) * lda + (k0 + cA0));
    tA1 = *(const v8bf*)(A  + (size_t)(blockM + rA1) * lda + (k0 + cA1));
    tB  = *(const v8bf*)(Bt + (size_t)(blockN + rB ) * ldb + (k0 + cB ));
  };
  loadTile(0);
#endif

  for (int ks = 0; ks < Ksteps; ++ks) {
    const int cur = ks & 1;

#if USE_ASYNC_LDS
    __builtin_amdgcn_s_wait_asynccnt(0);   // my async copies for tile `ks` landed
    __syncthreads();                       // everyone's copies landed
    if (ks + 1 < Ksteps) issueTile(ks + 1, cur ^ 1);
#else
    *(v8bf*)(&sA[cur][rA0 * LDSSTR + cA0]) = tA0;
    *(v8bf*)(&sA[cur][rA1 * LDSSTR + cA1]) = tA1;
    *(v8bf*)(&sB[cur][rB  * LDSSTR + cB ]) = tB;
    if (ks + 1 < Ksteps) loadTile(ks + 1); // global loads overlap with compute
    __syncthreads();
#endif
    // prefetch tile ks+2 toward L2 while tile ks is consumed
    if (ks + 2 < Ksteps) {
      __builtin_prefetch(A  + (size_t)(blockM + (tid & 127)) * lda + (ks * 32 + 64), 0, 1);
      __builtin_prefetch(Bt + (size_t)(blockN + (tid &  63)) * ldb + (ks * 32 + 64), 0, 1);
    }

    // --- build fragments (ISA 7.12.2 layouts) ---
    v16bf afrag[2], bfrag[2];
    #pragma unroll
    for (int mt = 0; mt < 2; ++mt) {
      int row = wm * 32 + mt * 16 + l15;     // M index
      int kb  = hsel * 8;                    // K block per lane half
      v8bf lo = *(const v8bf*)(&sA[cur][row * LDSSTR + kb]);
      v8bf hi = *(const v8bf*)(&sA[cur][row * LDSSTR + kb + 16]);
      afrag[mt] = __builtin_shufflevector(lo, hi, 0,1,2,3,4,5,6,7,8,9,10,11,12,13,14,15);
    }
    #pragma unroll
    for (int nt = 0; nt < 2; ++nt) {
      int row = wn * 32 + nt * 16 + l15;     // N index
      int kb  = hsel * 8;                    // K block per lane half
      v8bf lo = *(const v8bf*)(&sB[cur][row * LDSSTR + kb]);
      v8bf hi = *(const v8bf*)(&sB[cur][row * LDSSTR + kb + 16]);
      bfrag[nt] = __builtin_shufflevector(lo, hi, 0,1,2,3,4,5,6,7,8,9,10,11,12,13,14,15);
    }

    #pragma unroll
    for (int mt = 0; mt < 2; ++mt)
      #pragma unroll
      for (int nt = 0; nt < 2; ++nt)
        acc[mt][nt] = __builtin_amdgcn_wmma_f32_16x16x32_bf16(
            false, afrag[mt], false, bfrag[nt],
            (short)0, acc[mt][nt], false, false);
  }

  // --- epilogue ---
  float invL = 0.f, thr = 0.f;
  if (MODE == 2) {
    float L = *Lp;
    invL = 1.0f / L;
    thr  = COUPLE * invL;
  }

  #pragma unroll
  for (int mt = 0; mt < 2; ++mt) {
    #pragma unroll
    for (int nt = 0; nt < 2; ++nt) {
      int n = blockN + wn * 32 + nt * 16 + l15;
      if (n >= N) continue;
      #pragma unroll
      for (int r = 0; r < 8; ++r) {
        int   m = blockM + wm * 32 + mt * 16 + hsel * 8 + r;
        float g = acc[mt][nt][r];
        if (MODE == 1) {
          float val = g - Xres[(size_t)m * DS + n];
          OutBf[(size_t)m * DS_PAD + n] = (__bf16)val;
        } else if (MODE == 3) {
          OutF[(size_t)m * DS + n] = g;
        } else {
          size_t idx = (size_t)m * DD + n;
          float y  = Yf[idx];
          float zo = Zf[idx];
          float p  = y - g * invL;
          float a  = fabsf(p) - thr;
          float zn = (a > 0.f) ? copysignf(a, p) : 0.f;
          float yn = zn + mu * (zn - zo);
          Zf[idx] = zn;  Zbf[idx] = (__bf16)zn;
          Yf[idx] = yn;  Ybf[idx] = (__bf16)yn;
        }
      }
    }
  }
}

// ---------------------------------------------------------------------------
// Small helper kernels (init, bf16 conversion, power iteration)
// ---------------------------------------------------------------------------
__global__ void k_zero_u32(uint32_t* p, long n) {
  long i = (long)blockIdx.x * 256 + threadIdx.x;
  long s = (long)gridDim.x * 256;
  for (; i < n; i += s) p[i] = 0u;
}

__global__ void k_convD(const float* __restrict__ D, __bf16* __restrict__ Dbf) {
  int i = blockIdx.x * 256 + threadIdx.x;            // over DS_PAD * DD
  if (i >= DS_PAD * DD) return;
  int d = i / DD, k = i % DD;
  float v = (d < DS) ? D[(size_t)d * DD + k] : 0.f;
  Dbf[i] = (__bf16)v;
}

__global__ void k_convDt(const float* __restrict__ D, __bf16* __restrict__ Dt) {
  int i = blockIdx.x * 256 + threadIdx.x;            // over DD_PAD * DS_PAD
  if (i >= DD_PAD * DS_PAD) return;
  int k = i / DS_PAD, d = i % DS_PAD;
  float v = (k < DD && d < DS) ? D[(size_t)d * DD + k] : 0.f;
  Dt[i] = (__bf16)v;
}

__global__ void k_initv(float* v) {
  int k = blockIdx.x * 256 + threadIdx.x;
  if (k < DD) {
    unsigned h = (unsigned)k * 1103515245u + 12345u;
    h ^= h >> 13; h *= 2654435761u; h ^= h >> 16;
    v[k] = ((float)(h & 0xffffffu) / 8388608.0f) - 1.0f;
  }
}

// w[d] = dot(D[d,:], v)   -- one wave per row
__global__ __launch_bounds__(256) void k_mv_w(const float* __restrict__ D,
                                              const float* __restrict__ v,
                                              float* __restrict__ w) {
  int wid = threadIdx.x >> 5, lane = threadIdx.x & 31;
  int row = blockIdx.x * 8 + wid;
  if (row >= DS) return;
  float s = 0.f;
  for (int j = lane; j < DD; j += 32) s += D[(size_t)row * DD + j] * v[j];
  for (int o = 16; o > 0; o >>= 1) s += __shfl_down(s, o, 32);
  if (lane == 0) w[row] = s;
}

// u[k] = dot(D[:,k], w)   -- one wave per column
__global__ __launch_bounds__(256) void k_mv_u(const float* __restrict__ D,
                                              const float* __restrict__ w,
                                              float* __restrict__ u) {
  int wid = threadIdx.x >> 5, lane = threadIdx.x & 31;
  int k = blockIdx.x * 8 + wid;
  if (k >= DD) return;
  float s = 0.f;
  for (int d = lane; d < DS; d += 32) s += D[(size_t)d * DD + k] * w[d];
  for (int o = 16; o > 0; o >>= 1) s += __shfl_down(s, o, 32);
  if (lane == 0) u[k] = s;
}

// n = ||u||; L = n; v = u / n   -- single block, deterministic reduction
__global__ __launch_bounds__(256) void k_norm_scale(const float* __restrict__ u,
                                                    float* __restrict__ v,
                                                    float* __restrict__ Lp) {
  __shared__ float red[256];
  float s = 0.f;
  for (int i = threadIdx.x; i < DD; i += 256) s += u[i] * u[i];
  red[threadIdx.x] = s;
  __syncthreads();
  for (int o = 128; o > 0; o >>= 1) {
    if (threadIdx.x < o) red[threadIdx.x] += red[threadIdx.x + o];
    __syncthreads();
  }
  float n = sqrtf(red[0]);
  if (threadIdx.x == 0) *Lp = n;
  float inv = 1.0f / n;
  for (int i = threadIdx.x; i < DD; i += 256) v[i] = u[i] * inv;
}

// ---------------------------------------------------------------------------
// Orchestration
// ---------------------------------------------------------------------------
extern "C" void kernel_launch(void* const* d_in, const int* in_sizes, int n_in,
                              void* d_out, int out_size, void* d_ws, size_t ws_size,
                              hipStream_t stream) {
  (void)in_sizes; (void)n_in; (void)out_size; (void)ws_size;
  const float* X = (const float*)d_in[0];   // (2048, 1200) fp32
  const float* D = (const float*)d_in[1];   // (1200, 2400) fp32
  float* out = (float*)d_out;               // (2048, 1200) fp32

  // workspace layout (bytes)
  char* ws = (char*)d_ws;
  const size_t nY = (size_t)M_ROWS * DD;                       // 4,915,200
  float*  Yf   = (float*)(ws);                                 // f32
  float*  Zf   = (float*)(ws + nY * 4);                        // f32
  __bf16* Ybf  = (__bf16*)(ws + nY * 8);                       // bf16
  __bf16* Zbf  = (__bf16*)(ws + nY * 8 + nY * 2);
  __bf16* Rbf  = (__bf16*)(ws + nY * 8 + nY * 4);              // (2048, DS_PAD)
  size_t offDbf  = nY * 8 + nY * 4 + (size_t)M_ROWS * DS_PAD * 2;
  __bf16* Dbf  = (__bf16*)(ws + offDbf);                       // (DS_PAD, DD)
  size_t offDtbf = offDbf + (size_t)DS_PAD * DD * 2;
  __bf16* Dtbf = (__bf16*)(ws + offDtbf);                      // (DD_PAD, DS_PAD)
  size_t offPow  = offDtbf + (size_t)DD_PAD * DS_PAD * 2;
  float* pv = (float*)(ws + offPow);
  float* pw = pv + DD;
  float* pu = pw + DS;
  float* pL = pu + DD;

  // host-side Nesterov schedule (data independent, deterministic)
  float mus[ITER_FISTA];
  {
    float t = 1.0f;
    for (int i = 0; i < ITER_FISTA; ++i) {
      float tn = 0.5f * (1.0f + sqrtf(1.0f + 4.0f * t * t));
      mus[i] = (t - 1.0f) / tn;
      t = tn;
    }
  }

  // 1) zero the f32/bf16 state (Yf, Zf, Ybf, Zbf, Rbf incl. K pads)
  {
    long nbytes = (long)offDbf;
    long nu32 = nbytes / 4;
    k_zero_u32<<<4096, 256, 0, stream>>>((uint32_t*)ws, nu32);
  }
  // 2) bf16 dictionary copies (row-major and transposed, zero padded)
  k_convD <<<(DS_PAD * DD + 255) / 256, 256, 0, stream>>>(D, Dbf);
  k_convDt<<<(DD_PAD * DS_PAD + 255) / 256, 256, 0, stream>>>(D, Dtbf);
  // 3) power iteration for L (fp32)
  k_initv<<<(DD + 255) / 256, 256, 0, stream>>>(pv);
  for (int it = 0; it < ITER_EIG; ++it) {
    k_mv_w<<<(DS + 7) / 8, 256, 0, stream>>>(D, pv, pw);
    k_mv_u<<<(DD + 7) / 8, 256, 0, stream>>>(D, pw, pu);
    k_norm_scale<<<1, 256, 0, stream>>>(pu, pv, pL);
  }

  // 4) FISTA iterations: two WMMA GEMMs per step (launches = global barriers)
  dim3 blk(256);
  dim3 grid1((DS + 63) / 64, M_ROWS / 128);       // 19 x 16
  dim3 grid2(DD_PAD / 64,    M_ROWS / 128);       // 38 x 16
  for (int it = 0; it < ITER_FISTA; ++it) {
    // R = Y * D^T - X     (K = 2400)
    gemm_nt_wmma<1><<<grid1, blk, 0, stream>>>(
        Ybf, DD, Dbf, DD, DS, DD / 32,
        X, Rbf, nullptr, nullptr, nullptr, nullptr, nullptr, nullptr, 0.f);
    // G = R * D, fused soft-threshold + momentum   (K = 1216 incl. zero pad)
    gemm_nt_wmma<2><<<grid2, blk, 0, stream>>>(
        Rbf, DS_PAD, Dtbf, DS_PAD, DD, DS_PAD / 32,
        nullptr, nullptr, nullptr, pL, Zf, Zbf, Yf, Ybf, mus[it]);
  }

  // 5) final reconstruction from z:  out = Z * D^T
  gemm_nt_wmma<3><<<grid1, blk, 0, stream>>>(
      Zbf, DD, Dbf, DD, DS, DD / 32,
      nullptr, nullptr, out, nullptr, nullptr, nullptr, nullptr, nullptr, 0.f);
}
